// SwinTransformerV2Block_11596411699337
// MI455X (gfx1250) — compile-verified, matrix operand order checked by reading, not run on hardware
//
#include <hip/hip_runtime.h>
#include <cstdint>

// ---------- types ----------
typedef __attribute__((ext_vector_type(8)))  float        v8f;
typedef __attribute__((ext_vector_type(16))) __bf16       v16bf;
typedef __attribute__((ext_vector_type(4)))  unsigned int u32x4;
typedef __attribute__((ext_vector_type(8)))  unsigned int u32x8;

struct __align__(16) U4 { unsigned int x, y, z, w; };
struct __align__(8)  U2 { unsigned int x, y; };
struct __align__(16) F4 { float x, y, z, w; };

union Frag { v16bf v; U4 q[2]; };

__device__ __forceinline__ unsigned short f2bf(float f) {
  unsigned int u = __builtin_bit_cast(unsigned int, f);
  unsigned int r = u + 0x7FFFu + ((u >> 16) & 1u);   // round-to-nearest-even
  return (unsigned short)(r >> 16);
}

// region id for the shifted-window mask (H=W=32, WS=8, SS=4)
__device__ __forceinline__ int rid(int c) { return c < 24 ? 0 : (c < 28 ? 1 : 2); }

// ---------- TENSORcnt wait ----------
#if defined(__has_builtin)
#if __has_builtin(__builtin_amdgcn_s_wait_tensorcnt)
#define WAIT_TENSORCNT0() __builtin_amdgcn_s_wait_tensorcnt(0)
#endif
#endif
#ifndef WAIT_TENSORCNT0
#define WAIT_TENSORCNT0() asm volatile("s_wait_tensorcnt 0x0" ::: "memory")
#endif

// ---------- Tensor Data Mover: 2D bf16 tile (global -> LDS) ----------
// tile_x elems contiguous (data_size=2B), tile_y rows, row stride = stride_elems.
// D# group0/group1 built in SGPRs; 2-group form (2D tensor), no multicast.
__device__ __forceinline__ void tdm_load_tile_2d(unsigned lds_off, const void* gaddr,
                                                 unsigned tensor_x, unsigned tensor_y,
                                                 unsigned tile_x, unsigned tile_y,
                                                 unsigned stride_elems) {
  unsigned long long ga = (unsigned long long)(size_t)gaddr;
  u32x4 g0;
  g0[0] = 1u;                                              // count=1, user mode
  g0[1] = lds_off;                                         // lds_addr
  g0[2] = (unsigned)ga;                                    // global_addr[31:0]
  g0[3] = (unsigned)((ga >> 32) & 0x01FFFFFFu) | (2u << 30);  // addr[56:32] | type=2
  u32x8 g1;
  g1[0] = 1u << 16;                                        // data_size=2B, no mask/pad/iter
  g1[1] = (tensor_x & 0xFFFFu) << 16;                      // tensor_dim0[15:0] @bit48
  g1[2] = (tensor_x >> 16) | ((tensor_y & 0xFFFFu) << 16); // dim0[31:16] | dim1[15:0]
  g1[3] = (tensor_y >> 16) | (tile_x << 16);               // dim1[31:16] | tile_dim0
  g1[4] = tile_y;                                          // tile_dim1 (tile_dim2=0)
  g1[5] = stride_elems;                                    // tensor_dim0_stride[31:0]
  g1[6] = 0u;                                              // stride hi | dim1_stride lo
  g1[7] = 0u;
  asm volatile("tensor_load_to_lds %0, %1" :: "s"(g0), "s"(g1) : "memory");
}

// ---------- small precompute kernels ----------
__global__ void f2bf_kernel(const float* __restrict__ in, unsigned short* __restrict__ out, int n) {
  int i = blockIdx.x * 256 + threadIdx.x;
  if (i < n) out[i] = f2bf(in[i]);
}

__global__ void build_qkv_bias(const float* __restrict__ qb, const float* __restrict__ vb,
                               float* __restrict__ out) {
  int i = blockIdx.x * 256 + threadIdx.x;
  if (i < 1536) out[i] = (i < 512) ? qb[i] : ((i < 1024) ? 0.f : vb[i - 1024]);
}

__device__ __forceinline__ float relcoord(int idx) {
  float c = (float)(idx - 7);
  c = c / 7.f * 8.f;
  float v = log2f(fabsf(c) + 1.f) * (1.f / 3.f);   // /log2(8)
  return c < 0.f ? -v : v;
}

// tbl[i][h] = (relu(rel_table[i] @ w1^T + b1)) @ w2^T    (225 x 16)
__global__ __launch_bounds__(128) void cpb_tbl(const float* __restrict__ w1, const float* __restrict__ b1,
                                               const float* __restrict__ w2, float* __restrict__ tbl) {
  __shared__ float hid[512];
  int i  = blockIdx.x;           // 0..224
  int ai = i / 15, bi = i % 15;
  float t0 = relcoord(ai), t1 = relcoord(bi);
  for (int j = threadIdx.x; j < 512; j += 128) {
    float v = w1[j * 2] * t0 + w1[j * 2 + 1] * t1 + b1[j];
    hid[j] = v > 0.f ? v : 0.f;
  }
  __syncthreads();
  if (threadIdx.x < 16) {
    int h = threadIdx.x;
    float s = 0.f;
    for (int j = 0; j < 512; j++) s += hid[j] * w2[h * 512 + j];
    tbl[i * 16 + h] = s;
  }
}

// rpb[h][n][m] = 16*sigmoid(tbl[rel_idx(n,m)][h])   (16 x 64 x 64)
__global__ void rpb_build(const float* __restrict__ tbl, float* __restrict__ rpb) {
  int idx = blockIdx.x * 256 + threadIdx.x;          // 65536
  int h = idx >> 12, nm = idx & 4095, n = nm >> 6, m = nm & 63;
  int d0 = (n >> 3) - (m >> 3) + 7;
  int d1 = (n & 7)  - (m & 7)  + 7;
  float x = tbl[(d0 * 15 + d1) * 16 + h];
  rpb[idx] = 16.f / (1.f + __expf(-x));
}

// ---------- tiled WMMA GEMM: C(MxN) = A(MxK) @ W(NxK)^T + bias ----------
// Workgroup tile 128M x 64N, 4 waves; each wave: 32M x 64N = 2x4 WMMA tiles.
// B (bf16 weights) tile fetched by the Tensor Data Mover into LDS (TENSORcnt).
// AMODE 0: A fp32 row-major   1: swin window-gather from x(64,32,32,512)   2: A bf16 row-major
// EPI   0: none               1: exact GELU
// OUTBF 0: fp32 out           1: bf16 out
template <int AMODE, int EPI, int OUTBF>
__global__ __launch_bounds__(128) void gemm_wmma(const void* __restrict__ Aptr,
                                                 const unsigned short* __restrict__ Bw,
                                                 const float* __restrict__ bias,
                                                 void* __restrict__ Cptr,
                                                 int M, int N, int K) {
  __shared__ unsigned short As[128][32];   // 8 KB
  __shared__ unsigned short Bs[64][32];    // 4 KB
  const int tid = threadIdx.x;
  const int wv  = tid >> 5;
  const int ln  = tid & 31;
  const int tm0 = blockIdx.y * 128;
  const int tn0 = blockIdx.x * 64;

  v8f acc[2][4] = {};

  const int hs_ = ln >> 4;      // half-wave select
  const int rs  = ln & 15;      // row-in-tile
  const int kb  = hs_ * 8;      // k sub-offset

  const unsigned bs_off = (unsigned)(size_t)(&Bs[0][0]);   // LDS byte offset

  for (int kc = 0; kc < K; kc += 32) {
    // --- B tile via TDM (wave 0 issues; DMA overlaps the A-tile VALU load) ---
    if (tid < 32) {
      tdm_load_tile_2d(bs_off, Bw + (size_t)tn0 * K + kc,
                       (unsigned)K, (unsigned)N, 32u, 64u, (unsigned)K);
    }
    // --- load A tile (128x32) with fp32->bf16 convert (and swin gather for AMODE 1) ---
    for (int it = 0; it < 8; it++) {
      int row = it * 16 + (tid >> 3);
      int col = (tid & 7) * 4;
      if (AMODE == 2) {
        const unsigned short* Ab = (const unsigned short*)Aptr;
        U2 v = *(const U2*)(Ab + (size_t)(tm0 + row) * K + kc + col);
        *(unsigned int*)&As[row][col]     = v.x;
        *(unsigned int*)&As[row][col + 2] = v.y;
      } else {
        const float* src;
        if (AMODE == 1) {
          int m = tm0 + row, w = m >> 6, t = m & 63;
          int b = w >> 4, wi = w & 15;
          int hsx = (wi >> 2) * 8 + (t >> 3);
          int wsx = (wi & 3) * 8 + (t & 7);
          int sh = (hsx + 4) & 31, sw = (wsx + 4) & 31;       // roll(x, -4)
          src = (const float*)Aptr + (((size_t)(b * 32 + sh) * 32) + sw) * 512 + kc + col;
        } else {
          src = (const float*)Aptr + (size_t)(tm0 + row) * K + kc + col;
        }
        F4 f = *(const F4*)src;
        *(unsigned int*)&As[row][col]     = (unsigned int)f2bf(f.x) | ((unsigned int)f2bf(f.y) << 16);
        *(unsigned int*)&As[row][col + 2] = (unsigned int)f2bf(f.z) | ((unsigned int)f2bf(f.w) << 16);
      }
    }
    if (kc + 32 < K && AMODE != 1)
      __builtin_prefetch((const char*)Aptr + ((size_t)tm0 * K + kc + 32) * (AMODE == 2 ? 2 : 4), 0, 0);
    if (tid < 32) WAIT_TENSORCNT0();     // wave 0: TDM done before publishing
    __syncthreads();

    Frag a0, a1;
    a0.q[0] = *(const U4*)&As[wv * 32 + rs][kb];
    a0.q[1] = *(const U4*)&As[wv * 32 + rs][16 + kb];
    a1.q[0] = *(const U4*)&As[wv * 32 + 16 + rs][kb];
    a1.q[1] = *(const U4*)&As[wv * 32 + 16 + rs][16 + kb];
#pragma unroll
    for (int c = 0; c < 4; c++) {
      Frag b;
      b.q[0] = *(const U4*)&Bs[c * 16 + rs][kb];
      b.q[1] = *(const U4*)&Bs[c * 16 + rs][16 + kb];
      acc[0][c] = __builtin_amdgcn_wmma_f32_16x16x32_bf16(false, a0.v, false, b.v,
                                                          (short)0, acc[0][c], false, false);
      acc[1][c] = __builtin_amdgcn_wmma_f32_16x16x32_bf16(false, a1.v, false, b.v,
                                                          (short)0, acc[1][c], false, false);
    }
    __syncthreads();
  }

  // --- epilogue ---
#pragma unroll
  for (int s = 0; s < 2; s++) {
#pragma unroll
    for (int c = 0; c < 4; c++) {
      int n = tn0 + c * 16 + rs;
      float bn = bias ? bias[n] : 0.f;
#pragma unroll
      for (int r = 0; r < 8; r++) {
        int m = tm0 + wv * 32 + s * 16 + hs_ * 8 + r;
        float v = acc[s][c][r] + bn;
        if (EPI == 1) v = 0.5f * v * (1.f + erff(v * 0.70710678118f));
        if (OUTBF) ((unsigned short*)Cptr)[(size_t)m * N + n] = f2bf(v);
        else       ((float*)Cptr)[(size_t)m * N + n] = v;
      }
    }
  }
}

// ---------- fused window attention: one workgroup per (window, head) ----------
__global__ __launch_bounds__(128) void attn_kernel(const float* __restrict__ qkv,
                                                   const float* __restrict__ rpb,
                                                   const float* __restrict__ logit_scale,
                                                   float* __restrict__ attn_out) {
  __shared__ unsigned short qn[64][32];
  __shared__ unsigned short kn[64][32];
  __shared__ unsigned short vt[32][64];
  __shared__ float          Sf[64][64];
  __shared__ unsigned short Pb[64][64];
  __shared__ float          s_scale;

  const int w   = blockIdx.x;     // window 0..1023
  const int h   = blockIdx.y;     // head 0..15
  const int tid = threadIdx.x;
  const int wv  = tid >> 5;
  const int ln  = tid & 31;
  const int hs_ = ln >> 4, rs = ln & 15, kb = hs_ * 8;

  if (tid == 0) s_scale = __expf(fminf(logit_scale[h], 4.605170186f));   // log(100)

  // load + l2-normalize q,k (bf16 into LDS), load v transposed
  {
    int row = tid >> 1;
    int hd  = (tid & 1) * 16;
    const float* qrow = qkv + (size_t)(w * 64 + row) * 1536 + h * 32;
    float qv[16], kv[16];
    float ssq = 0.f, ssk = 0.f;
#pragma unroll
    for (int j = 0; j < 16; j++) {
      qv[j] = qrow[hd + j];        ssq += qv[j] * qv[j];
      kv[j] = qrow[512 + hd + j];  ssk += kv[j] * kv[j];
    }
    ssq += __shfl_xor(ssq, 1);
    ssk += __shfl_xor(ssk, 1);
    float iq = 1.f / fmaxf(sqrtf(ssq), 1e-12f);
    float ik = 1.f / fmaxf(sqrtf(ssk), 1e-12f);
#pragma unroll
    for (int j = 0; j < 16; j++) {
      qn[row][hd + j] = f2bf(qv[j] * iq);
      kn[row][hd + j] = f2bf(kv[j] * ik);
      vt[hd + j][row] = f2bf(qrow[1024 + hd + j]);
    }
  }
  __syncthreads();

  // S = qn @ kn^T  (wave wv owns rows 16*wv..+15)
  {
    v8f acc[4] = {};
    Frag a;
    a.q[0] = *(const U4*)&qn[wv * 16 + rs][kb];
    a.q[1] = *(const U4*)&qn[wv * 16 + rs][16 + kb];
#pragma unroll
    for (int c = 0; c < 4; c++) {
      Frag b;
      b.q[0] = *(const U4*)&kn[c * 16 + rs][kb];
      b.q[1] = *(const U4*)&kn[c * 16 + rs][16 + kb];
      acc[c] = __builtin_amdgcn_wmma_f32_16x16x32_bf16(false, a.v, false, b.v,
                                                       (short)0, acc[c], false, false);
    }
#pragma unroll
    for (int c = 0; c < 4; c++)
#pragma unroll
      for (int r = 0; r < 8; r++)
        Sf[wv * 16 + hs_ * 8 + r][c * 16 + rs] = acc[c][r];
  }
  __syncthreads();

  // masked+biased softmax, rows owned by threads 0..63
  if (tid < 64) {
    int n  = tid;
    int wi = w & 15;
    int nhs = (wi >> 2) * 8 + (n >> 3), nws = (wi & 3) * 8 + (n & 7);
    int cn = rid(nhs) * 3 + rid(nws);
    float scale = s_scale;
    float mx = -1e30f;
    for (int m = 0; m < 64; m++) {
      int mhs = (wi >> 2) * 8 + (m >> 3), mws = (wi & 3) * 8 + (m & 7);
      int cm = rid(mhs) * 3 + rid(mws);
      float val = Sf[n][m] * scale + rpb[(size_t)h * 4096 + n * 64 + m] + (cn != cm ? -100.f : 0.f);
      Sf[n][m] = val;
      mx = fmaxf(mx, val);
    }
    float s = 0.f;
    for (int m = 0; m < 64; m++) {
      float e = __expf(Sf[n][m] - mx);
      s += e;
      Sf[n][m] = e;
    }
    float inv = 1.f / s;
    for (int m = 0; m < 64; m++) Pb[n][m] = f2bf(Sf[n][m] * inv);
  }
  __syncthreads();

  // O = P @ V   (K = 64 in two 32-chunks; two 16-wide d tiles)
  v8f o[2] = {};
#pragma unroll
  for (int kc = 0; kc < 64; kc += 32) {
    Frag a;
    a.q[0] = *(const U4*)&Pb[wv * 16 + rs][kc + kb];
    a.q[1] = *(const U4*)&Pb[wv * 16 + rs][kc + 16 + kb];
#pragma unroll
    for (int d = 0; d < 2; d++) {
      Frag b;
      b.q[0] = *(const U4*)&vt[d * 16 + rs][kc + kb];
      b.q[1] = *(const U4*)&vt[d * 16 + rs][kc + 16 + kb];
      o[d] = __builtin_amdgcn_wmma_f32_16x16x32_bf16(false, a.v, false, b.v,
                                                     (short)0, o[d], false, false);
    }
  }

  // scatter-store with window reverse + roll(+4,+4) fused
  {
    int b_ = w >> 4, wi = w & 15;
#pragma unroll
    for (int d = 0; d < 2; d++)
#pragma unroll
      for (int r = 0; r < 8; r++) {
        int t = wv * 16 + hs_ * 8 + r;
        int dcol = d * 16 + rs;
        int hsr = (wi >> 2) * 8 + (t >> 3);
        int wsr = (wi & 3) * 8 + (t & 7);
        int oh = (hsr + 4) & 31, ow = (wsr + 4) & 31;
        attn_out[(((size_t)(b_ * 32 + oh) * 32) + ow) * 512 + h * 32 + dcol] = o[d][r];
      }
  }
}

// ---------- out = res + LayerNorm(src)*g + b   (one 128-thread block per 512-row) ----------
__global__ __launch_bounds__(128) void ln_residual(const float* __restrict__ src,
                                                   const float* __restrict__ res,
                                                   const float* __restrict__ g,
                                                   const float* __restrict__ bvec,
                                                   float* __restrict__ out) {
  __shared__ float red[128];
  const size_t row = blockIdx.x;
  const int tid = threadIdx.x;
  const float* s = src + row * 512;
  float x[4];
  float sum = 0.f;
#pragma unroll
  for (int j = 0; j < 4; j++) { x[j] = s[tid + j * 128]; sum += x[j]; }
  red[tid] = sum; __syncthreads();
  for (int off = 64; off > 0; off >>= 1) {
    if (tid < off) red[tid] += red[tid + off];
    __syncthreads();
  }
  float mu = red[0] * (1.f / 512.f);
  __syncthreads();
  float vs = 0.f;
#pragma unroll
  for (int j = 0; j < 4; j++) { float d = x[j] - mu; vs += d * d; }
  red[tid] = vs; __syncthreads();
  for (int off = 64; off > 0; off >>= 1) {
    if (tid < off) red[tid] += red[tid + off];
    __syncthreads();
  }
  float inv = rsqrtf(red[0] * (1.f / 512.f) + 1e-5f);
#pragma unroll
  for (int j = 0; j < 4; j++) {
    int c = tid + j * 128;
    out[row * 512 + c] = res[row * 512 + c] + (x[j] - mu) * inv * g[c] + bvec[c];
  }
}

// ---------- host ----------
extern "C" void kernel_launch(void* const* d_in, const int* in_sizes, int n_in,
                              void* d_out, int out_size, void* d_ws, size_t ws_size,
                              hipStream_t stream) {
  const float* x           = (const float*)d_in[0];
  const float* qkv_w       = (const float*)d_in[1];
  const float* q_bias      = (const float*)d_in[2];
  const float* v_bias      = (const float*)d_in[3];
  const float* logit_scale = (const float*)d_in[4];
  const float* cpb_w1      = (const float*)d_in[5];
  const float* cpb_b1      = (const float*)d_in[6];
  const float* cpb_w2      = (const float*)d_in[7];
  const float* proj_w      = (const float*)d_in[8];
  const float* proj_b      = (const float*)d_in[9];
  const float* norm1_g     = (const float*)d_in[10];
  const float* norm1_b     = (const float*)d_in[11];
  const float* fc1_w       = (const float*)d_in[12];
  const float* fc1_b       = (const float*)d_in[13];
  const float* fc2_w       = (const float*)d_in[14];
  const float* fc2_b       = (const float*)d_in[15];
  const float* norm2_g     = (const float*)d_in[16];
  const float* norm2_b     = (const float*)d_in[17];

  const size_t M = 65536;                 // 64 * 32*32 tokens
  char* p = (char*)d_ws;
  float* qkvbuf = (float*)p;          p += M * 1536 * sizeof(float);
  float* attn_o = (float*)p;          p += M * 512 * sizeof(float);
  float* x1     = (float*)p;          p += M * 512 * sizeof(float);
  float* mbuf   = (float*)p;          p += M * 512 * sizeof(float);
  unsigned short* hbuf  = (unsigned short*)p; p += M * 2048 * sizeof(unsigned short);
  unsigned short* wqkv  = (unsigned short*)p; p += (size_t)1536 * 512 * 2;
  unsigned short* wproj = (unsigned short*)p; p += (size_t)512 * 512 * 2;
  unsigned short* wfc1  = (unsigned short*)p; p += (size_t)2048 * 512 * 2;
  unsigned short* wfc2  = (unsigned short*)p; p += (size_t)512 * 2048 * 2;
  float* qkvbias = (float*)p;         p += 1536 * sizeof(float);
  float* tbl     = (float*)p;         p += 225 * 16 * sizeof(float);
  float* rpbbuf  = (float*)p;         p += 16 * 4096 * sizeof(float);

  // weights -> bf16
  f2bf_kernel<<<(1536 * 512 + 255) / 256, 256, 0, stream>>>(qkv_w, wqkv, 1536 * 512);
  f2bf_kernel<<<(512 * 512 + 255) / 256, 256, 0, stream>>>(proj_w, wproj, 512 * 512);
  f2bf_kernel<<<(2048 * 512 + 255) / 256, 256, 0, stream>>>(fc1_w, wfc1, 2048 * 512);
  f2bf_kernel<<<(512 * 2048 + 255) / 256, 256, 0, stream>>>(fc2_w, wfc2, 512 * 2048);
  build_qkv_bias<<<6, 256, 0, stream>>>(q_bias, v_bias, qkvbias);
  cpb_tbl<<<225, 128, 0, stream>>>(cpb_w1, cpb_b1, cpb_w2, tbl);
  rpb_build<<<256, 256, 0, stream>>>(tbl, rpbbuf);

  // QKV projection with fused shift+window gather (TDM-fed B tiles)
  gemm_wmma<1, 0, 0><<<dim3(1536 / 64, M / 128), 128, 0, stream>>>(x, wqkv, qkvbias, qkvbuf,
                                                                   (int)M, 1536, 512);
  // attention per (window, head) with fused reverse-shift scatter
  attn_kernel<<<dim3(1024, 16), 128, 0, stream>>>(qkvbuf, rpbbuf, logit_scale, attn_o);

  // proj -> mbuf ; x1 = x + LN(mbuf)
  gemm_wmma<0, 0, 0><<<dim3(512 / 64, M / 128), 128, 0, stream>>>(attn_o, wproj, proj_b, mbuf,
                                                                  (int)M, 512, 512);
  ln_residual<<<M, 128, 0, stream>>>(mbuf, x, norm1_g, norm1_b, x1);

  // fc1 (GELU, bf16 out) ; fc2 ; out = x1 + LN(m)
  gemm_wmma<0, 1, 1><<<dim3(2048 / 64, M / 128), 128, 0, stream>>>(x1, wfc1, fc1_b, hbuf,
                                                                   (int)M, 2048, 512);
  gemm_wmma<2, 0, 0><<<dim3(512 / 64, M / 128), 128, 0, stream>>>(hbuf, wfc2, fc2_b, mbuf,
                                                                  (int)M, 512, 2048);
  ln_residual<<<M, 128, 0, stream>>>(mbuf, x1, norm2_g, norm2_b, (float*)d_out);
}
